// _TimeMixing_3238405341544
// MI455X (gfx1250) — matrix operand order repaired and checked by reading, not compiled, not run on hardware
//
#include <hip/hip_runtime.h>
#include <math.h>

// ---------------------------------------------------------------------------
// TimeMixing: y = relu( ((x-mean)*rsqrt(var+eps)*gamma+beta) @ W^T + b ) + x
// x:[32,512,1024] f32, W:[1024,1024], b:[1024], scalar gamma/beta.
// M = 32*512 = 16384, N = K = 1024.
// ---------------------------------------------------------------------------

typedef __attribute__((ext_vector_type(16))) _Float16 v16h;
typedef __attribute__((ext_vector_type(8)))  _Float16 h8;
typedef __attribute__((ext_vector_type(4)))  _Float16 h4;
typedef __attribute__((ext_vector_type(8)))  float    v8f;
typedef __attribute__((ext_vector_type(4)))  float    v4f;

#define M_TOTAL 16384
#define T_DIM   1024
#define N_ELEM  16777216ULL   // 32*512*1024

// workspace layout (bytes); total ~35.7 MB
#define WS_STATS 0                        // 2 floats: mean, invstd
#define WS_PART  256                      // 2048 * 2 floats
#define WS_WH    32768                    // 1024*1024 halves = 2 MB
#define WS_XH    (32768 + 2*1024*1024)    // 16384*1024 halves = 32 MB

// ---------------- kernel 1: per-block partial sums (deterministic) ----------
__global__ __launch_bounds__(256) void k_reduce(const float* __restrict__ x,
                                                float* __restrict__ partials) {
  __shared__ float sS[256];
  __shared__ float sQ[256];
  const int t = threadIdx.x;
  const size_t base = (size_t)blockIdx.x * 8192 + (size_t)t * 4;
  float s = 0.f, q = 0.f;
#pragma unroll
  for (int i = 0; i < 8; ++i) {
    v4f v = *(const v4f*)(x + base + (size_t)i * 1024);
#pragma unroll
    for (int e = 0; e < 4; ++e) { s += v[e]; q += v[e] * v[e]; }
  }
  sS[t] = s; sQ[t] = q;
  __syncthreads();
  for (int off = 128; off > 0; off >>= 1) {
    if (t < off) { sS[t] += sS[t + off]; sQ[t] += sQ[t + off]; }
    __syncthreads();
  }
  if (t == 0) { partials[2 * blockIdx.x] = sS[0]; partials[2 * blockIdx.x + 1] = sQ[0]; }
}

// ---------------- kernel 2: finalize mean / invstd in f64 -------------------
__global__ __launch_bounds__(256) void k_finalize(const float* __restrict__ partials,
                                                  float* __restrict__ stats) {
  __shared__ double sS[256];
  __shared__ double sQ[256];
  const int t = threadIdx.x;
  double s = 0.0, q = 0.0;
  for (int i = t; i < 2048; i += 256) {
    s += (double)partials[2 * i];
    q += (double)partials[2 * i + 1];
  }
  sS[t] = s; sQ[t] = q;
  __syncthreads();
  for (int off = 128; off > 0; off >>= 1) {
    if (t < off) { sS[t] += sS[t + off]; sQ[t] += sQ[t + off]; }
    __syncthreads();
  }
  if (t == 0) {
    const double n = (double)N_ELEM;
    const double mean = sS[0] / n;
    const double var  = sQ[0] / n - mean * mean;
    stats[0] = (float)mean;
    stats[1] = (float)(1.0 / sqrt(var + 1e-5));
  }
}

// -------- kernel 3: fused normalize + affine + f32->f16 (A operand) ---------
__global__ __launch_bounds__(256) void k_norm(const float* __restrict__ x,
                                              const float* __restrict__ stats,
                                              const float* __restrict__ gamma,
                                              const float* __restrict__ beta,
                                              _Float16* __restrict__ xh) {
  const float g = gamma[0] * stats[1];
  const float c = beta[0] - stats[0] * g;   // y = x*g + c
  const size_t i = ((size_t)blockIdx.x * 256 + threadIdx.x) * 4;
  v4f v = *(const v4f*)(x + i);
  h4 h;
#pragma unroll
  for (int e = 0; e < 4; ++e) h[e] = (_Float16)(v[e] * g + c);
  *(h4*)(xh + i) = h;
}

// ---------------- kernel 4: W f32 -> f16 ------------------------------------
__global__ __launch_bounds__(256) void k_cvtw(const float* __restrict__ W,
                                              _Float16* __restrict__ wh) {
  const size_t i = ((size_t)blockIdx.x * 256 + threadIdx.x) * 4;
  v4f v = *(const v4f*)(W + i);
  h4 h;
#pragma unroll
  for (int e = 0; e < 4; ++e) h[e] = (_Float16)v[e];
  *(h4*)(wh + i) = h;
}

// ---------------- kernel 5: WMMA GEMM + bias + relu + residual --------------
#define BM 128
#define BN 128
#define BK 32
#define ROWH 40        // padded halves/row (80B stride -> conflict-free b128)
#define NKSTEP 32      // 1024 / 32

__device__ __forceinline__ void async_b128(unsigned lds_off, const void* gaddr) {
  // GLOBAL_LOAD_ASYNC_TO_LDS_B128 (ASYNCcnt-tracked DMA, gfx1250)
  asm volatile("global_load_async_to_lds_b128 %0, %1, off"
               :: "v"(lds_off), "v"(gaddr) : "memory");
}
__device__ __forceinline__ void wait_async0() {
  asm volatile("s_wait_asynccnt 0x0" ::: "memory");
}
__device__ __forceinline__ v16h mk16(h8 lo, h8 hh) {
  return __builtin_shufflevector(lo, hh, 0,1,2,3,4,5,6,7,8,9,10,11,12,13,14,15);
}

__global__ __launch_bounds__(256) void k_gemm(const _Float16* __restrict__ xh,
                                              const _Float16* __restrict__ wh,
                                              const float* __restrict__ x,
                                              const float* __restrict__ bias,
                                              float* __restrict__ out) {
  __shared__ _Float16 sA[2][BM * ROWH];
  __shared__ _Float16 sB[2][BN * ROWH];

  const int t    = threadIdx.x;
  const int lane = t & 31;
  const int ln   = lane & 15;
  const int hi   = lane >> 4;
  const int w    = t >> 5;        // 8 waves
  const int wm   = w >> 1;        // 0..3  (M direction, 32 rows each)
  const int wn   = w & 1;         // 0..1  (N direction, 64 cols each)
  const size_t mBlk = (size_t)blockIdx.y * BM;
  const size_t nBlk = (size_t)blockIdx.x * BN;

  // each thread loads 2 x 16B chunks per tile: chunk c -> row c>>2, seg (c&3)*8 halves
  const int c1 = t + 256;
  const int r0 = t >> 2,  s0 = (t & 3) * 8;
  const int r1 = c1 >> 2, s1 = (c1 & 3) * 8;

  auto issue = [&](int b, int k0) {
    async_b128((unsigned)(size_t)&sA[b][r0 * ROWH + s0], xh + (mBlk + r0) * T_DIM + k0 + s0);
    async_b128((unsigned)(size_t)&sA[b][r1 * ROWH + s1], xh + (mBlk + r1) * T_DIM + k0 + s1);
    async_b128((unsigned)(size_t)&sB[b][r0 * ROWH + s0], wh + (nBlk + r0) * T_DIM + k0 + s0);
    async_b128((unsigned)(size_t)&sB[b][r1 * ROWH + s1], wh + (nBlk + r1) * T_DIM + k0 + s1);
  };

  v8f acc[2][4] = {};

  issue(0, 0);
  wait_async0();
  __syncthreads();

  int buf = 0;
  for (int kt = 0; kt < NKSTEP; ++kt) {
    if (kt + 1 < NKSTEP) issue(buf ^ 1, (kt + 1) * BK);   // overlap DMA with WMMA

    v16h Af[2], Bf[4];
#pragma unroll
    for (int i = 0; i < 2; ++i) {
      const int row = wm * 32 + i * 16 + ln;
      h8 lo = *(const h8*)(&sA[buf][row * ROWH + hi * 8]);        // K = 8*hi + 0..7
      h8 hh = *(const h8*)(&sA[buf][row * ROWH + 16 + hi * 8]);   // K = 16 + 8*hi + 0..7
      Af[i] = mk16(lo, hh);
    }
#pragma unroll
    for (int j = 0; j < 4; ++j) {
      const int row = wn * 64 + j * 16 + ln;                      // row u of W
      h8 lo = *(const h8*)(&sB[buf][row * ROWH + hi * 8]);        // K = 8*hi + 0..7
      h8 hh = *(const h8*)(&sB[buf][row * ROWH + 16 + hi * 8]);   // K = 16 + 8*hi + 0..7
      Bf[j] = mk16(lo, hh);
    }
#pragma unroll
    for (int i = 0; i < 2; ++i)
#pragma unroll
      for (int j = 0; j < 4; ++j)
        acc[i][j] = __builtin_amdgcn_wmma_f32_16x16x32_f16(
            false, Af[i], false, Bf[j], (short)0, acc[i][j], false, false);

    wait_async0();
    __syncthreads();
    buf ^= 1;
  }

  // epilogue: bias + relu + residual; C/D layout: lane ln = col, row = r + 8*hi
#pragma unroll
  for (int j = 0; j < 4; ++j) {
    const size_t u  = nBlk + (size_t)(wn * 64 + j * 16 + ln);
    const float  bj = bias[u];
#pragma unroll
    for (int i = 0; i < 2; ++i) {
      const size_t mBase = mBlk + (size_t)(wm * 32 + i * 16 + hi * 8);
#pragma unroll
      for (int r = 0; r < 8; ++r) {
        const size_t idx = (mBase + r) * T_DIM + u;
        float v = acc[i][j][r] + bj;
        v = v > 0.f ? v : 0.f;
        out[idx] = v + x[idx];
      }
    }
  }
}

// ---------------------------------------------------------------------------
extern "C" void kernel_launch(void* const* d_in, const int* in_sizes, int n_in,
                              void* d_out, int out_size, void* d_ws, size_t ws_size,
                              hipStream_t stream) {
  const float* x     = (const float*)d_in[0];
  const float* W     = (const float*)d_in[1];
  const float* b     = (const float*)d_in[2];
  const float* gamma = (const float*)d_in[3];
  const float* beta  = (const float*)d_in[4];
  float* out = (float*)d_out;

  char* ws = (char*)d_ws;
  float*     stats    = (float*)(ws + WS_STATS);
  float*     partials = (float*)(ws + WS_PART);
  _Float16*  wh       = (_Float16*)(ws + WS_WH);
  _Float16*  xh       = (_Float16*)(ws + WS_XH);

  k_reduce  <<<2048,  256, 0, stream>>>(x, partials);
  k_finalize<<<1,     256, 0, stream>>>(partials, stats);
  k_norm    <<<16384, 256, 0, stream>>>(x, stats, gamma, beta, xh);
  k_cvtw    <<<1024,  256, 0, stream>>>(W, wh);

  dim3 grid(T_DIM / BN, M_TOTAL / BM);   // (8, 128)
  k_gemm<<<grid, 256, 0, stream>>>(xh, wh, x, b, out);
}
